// M50_Func_14336600834650
// MI455X (gfx1250) — compile-verified, matrix operand order checked by reading, not run on hardware
//
#include <hip/hip_runtime.h>

typedef __attribute__((ext_vector_type(16))) _Float16 v16h;
typedef __attribute__((ext_vector_type(8)))  float    v8f;

// ---- fast transcendentals (v_exp_f32 + v_rcp_f32) ----
__device__ __forceinline__ float fast_tanh(float x) {
    // tanh(x) = 1 - 2/(exp(2x)+1); saturates correctly at +-inf
    float e = __expf(2.0f * x);
    return 1.0f - 2.0f * __builtin_amdgcn_rcpf(e + 1.0f);
}
__device__ __forceinline__ float smooth_step(float x) {
    return 0.5f * fast_tanh(5.0f * x) + 0.5f;
}

// LDS weight-pool offsets (floats); all float4-aligned
#define OFF_ETW1 0      // 48
#define OFF_ETB1 48     // 16
#define OFF_ETW2 64     // 256
#define OFF_ETB2 320    // 16
#define OFF_ETW3 336    // 16
#define OFF_QW1  352    // 32
#define OFF_QB1  384    // 16
#define OFF_QW2  400    // 256
#define OFF_QB2  656    // 16
#define OFF_QW3  672    // 16
#define OFF_B3S  688    // 2
#define SW_LEN   704

// One 16-wide MLP evaluated for 32 points held one-per-lane by this wave.
// Hidden 16x16 GEMM runs on the matrix pipe as two 16x16x32 f16 WMMAs.
template <int NIN>
__device__ __forceinline__ float mlp16(
    float x0, float x1, float x2,
    const float* __restrict__ w1,   // LDS: NIN x 16 row-major
    const float* __restrict__ b1,   // LDS: 16
    v16h bfrag,                     // hidden-weight B fragment (per-lane)
    float cbias,                    // b2[n] for this lane's column
    const float* __restrict__ w3,   // LDS: 16
    float b3,
    uint4* __restrict__ astWave,    // LDS staging: [32 pts][2 x uint4] (16 f16/row)
    float* __restrict__ hstWave,    // LDS staging: [32 pts][16 f32]
    int lane)
{
    const int ncol = lane & 15;
    const int hsel = lane >> 4;

    // ---- input layer (VALU): h1 = tanh(x @ W1 + b1), packed to f16 ----
    union { _Float16 h[16]; uint4 q[2]; } row;
    #pragma unroll
    for (int j4 = 0; j4 < 4; ++j4) {
        float4 acc = ((const float4*)b1)[j4];
        {
            float4 w = ((const float4*)(w1 + 0))[j4];
            acc.x = fmaf(x0, w.x, acc.x); acc.y = fmaf(x0, w.y, acc.y);
            acc.z = fmaf(x0, w.z, acc.z); acc.w = fmaf(x0, w.w, acc.w);
        }
        {
            float4 w = ((const float4*)(w1 + 16))[j4];
            acc.x = fmaf(x1, w.x, acc.x); acc.y = fmaf(x1, w.y, acc.y);
            acc.z = fmaf(x1, w.z, acc.z); acc.w = fmaf(x1, w.w, acc.w);
        }
        if (NIN == 3) {
            float4 w = ((const float4*)(w1 + 32))[j4];
            acc.x = fmaf(x2, w.x, acc.x); acc.y = fmaf(x2, w.y, acc.y);
            acc.z = fmaf(x2, w.z, acc.z); acc.w = fmaf(x2, w.w, acc.w);
        }
        row.h[4*j4+0] = (_Float16)fast_tanh(acc.x);
        row.h[4*j4+1] = (_Float16)fast_tanh(acc.y);
        row.h[4*j4+2] = (_Float16)fast_tanh(acc.z);
        row.h[4*j4+3] = (_Float16)fast_tanh(acc.w);
    }
    astWave[lane*2 + 0] = row.q[0];
    astWave[lane*2 + 1] = row.q[1];
    asm volatile("s_wait_dscnt 0" ::: "memory");   // same-wave LDS transpose RAW

    // ---- hidden layer on matrix pipe: two 16-point tiles ----
    union { v16h v; uint4 q[2]; } af;
    af.q[1].x = 0u; af.q[1].y = 0u; af.q[1].z = 0u; af.q[1].w = 0u; // K=16..31 zero
    v8f cc;
    #pragma unroll
    for (int r = 0; r < 8; ++r) cc[r] = cbias;     // C[m][n] = b2[n]

    #pragma unroll
    for (int tt = 0; tt < 2; ++tt) {
        // ISA A-layout: lanes 0-15 carry K0-7 of row M, lanes 16-31 carry K8-15
        af.q[0] = astWave[(tt*16 + ncol)*2 + hsel];
        v8f d = __builtin_amdgcn_wmma_f32_16x16x32_f16(
            false, af.v, false, bfrag, (short)0, cc, false, false);
        const int mb = tt*16 + hsel*8;             // D rows owned by this lane
        #pragma unroll
        for (int r = 0; r < 8; ++r)
            hstWave[(mb + r)*16 + ncol] = fast_tanh(d[r]);
    }
    asm volatile("s_wait_dscnt 0" ::: "memory");

    // ---- output layer: per-lane 16-wide dot with W3 ----
    float acc = b3;
    const float4* hr = (const float4*)(hstWave + lane*16);
    const float4* wv = (const float4*)w3;
    #pragma unroll
    for (int j4 = 0; j4 < 4; ++j4) {
        float4 h = hr[j4], w = wv[j4];
        acc = fmaf(h.x, w.x, acc); acc = fmaf(h.y, w.y, acc);
        acc = fmaf(h.z, w.z, acc); acc = fmaf(h.w, w.w, acc);
    }
    return acc;
}

__global__ __launch_bounds__(256) void m50_rhs_kernel(
    const float* __restrict__ t_in,
    const float* __restrict__ ssnow_in,
    const float* __restrict__ swater_in,
    const float* __restrict__ precp_s,
    const float* __restrict__ temp_s,
    const float* __restrict__ lday_s,
    const float* __restrict__ etW1, const float* __restrict__ etb1,
    const float* __restrict__ etW2, const float* __restrict__ etb2,
    const float* __restrict__ etW3, const float* __restrict__ etb3,
    const float* __restrict__ qW1,  const float* __restrict__ qb1,
    const float* __restrict__ qW2,  const float* __restrict__ qb2,
    const float* __restrict__ qW3,  const float* __restrict__ qb3,
    const float* __restrict__ Df_p, const float* __restrict__ Tmax_p,
    const float* __restrict__ Tmin_p,
    float* __restrict__ out, int n, int Tlen)
{
    __shared__ __align__(16) float sw[SW_LEN];
    __shared__ uint4 Ast[8][64];        // per-wave f16 A staging (32 pts x 16 halfs)
    __shared__ __align__(16) float Hst[8][32*16];   // per-wave f32 h2 staging

    const int tid = threadIdx.x;

    // cooperative weight load into LDS (divergent trip counts reconverge at barrier)
    sw[OFF_ETW2 + tid] = etW2[tid];
    sw[OFF_QW2  + tid] = qW2[tid];
    if (tid < 48) sw[OFF_ETW1 + tid] = etW1[tid];
    if (tid < 32) sw[OFF_QW1  + tid] = qW1[tid];
    if (tid < 16) {
        sw[OFF_ETB1 + tid] = etb1[tid];
        sw[OFF_ETB2 + tid] = etb2[tid];
        sw[OFF_ETW3 + tid] = etW3[tid];
        sw[OFF_QB1  + tid] = qb1[tid];
        sw[OFF_QB2  + tid] = qb2[tid];
        sw[OFF_QW3  + tid] = qW3[tid];
    }
    if (tid == 0) { sw[OFF_B3S] = etb3[0]; sw[OFF_B3S+1] = qb3[0]; }
    __syncthreads();

    const int lane = tid & 31;
    const int wav  = tid >> 5;
    const int ncol = lane & 15;
    const int hsel = lane >> 4;

    // persistent B fragments (ISA B-layout: lanes 0-15 = K0-7 of column N,
    // lanes 16-31 = K8-15; K16-31 zero-padded)
    union { v16h v; _Float16 h[16]; } etB, qB;
    #pragma unroll
    for (int j = 0; j < 8; ++j) {
        etB.h[j]   = (_Float16)sw[OFF_ETW2 + (hsel*8 + j)*16 + ncol];
        qB.h[j]    = (_Float16)sw[OFF_QW2  + (hsel*8 + j)*16 + ncol];
        etB.h[j+8] = (_Float16)0.0f;
        qB.h[j+8]  = (_Float16)0.0f;
    }
    const float etCb = sw[OFF_ETB2 + ncol];
    const float qCb  = sw[OFF_QB2  + ncol];
    const float etB3 = sw[OFF_B3S];
    const float qB3  = sw[OFF_B3S + 1];

    // ---- per-lane point setup: loads + uniform-grid interp ----
    const int i  = blockIdx.x * 256 + tid;
    const int ii = (i < n) ? i : (n - 1);           // branchless tail clamp
    const float tq      = t_in[ii];
    const float S_snow  = ssnow_in[ii];
    const float S_water = swater_in[ii];

    int k = (int)floorf(tq);
    k = (k < 0) ? 0 : ((k > Tlen - 2) ? (Tlen - 2) : k);
    const float fr = tq - (float)k;
    const float p0 = precp_s[k], p1 = precp_s[k+1];
    const float m0 = temp_s[k],  m1 = temp_s[k+1];
    const float l0 = lday_s[k],  l1 = lday_s[k+1];
    const float precp = fmaf(fr, p1 - p0, p0);
    const float temp  = fmaf(fr, m1 - m0, m0);
    const float lday  = fmaf(fr, l1 - l0, l0);

    // ---- the two MLPs (hidden layers on WMMA) ----
    float ET = mlp16<3>(S_snow, S_water, temp,
                        sw + OFF_ETW1, sw + OFF_ETB1, etB.v, etCb,
                        sw + OFF_ETW3, etB3, &Ast[wav][0], &Hst[wav][0], lane);
    float Q  = mlp16<2>(S_water, precp, 0.0f,
                        sw + OFF_QW1, sw + OFF_QB1, qB.v, qCb,
                        sw + OFF_QW3, qB3, &Ast[wav][0], &Hst[wav][0], lane);

    // ---- bucket physics ----
    const float Df = Df_p[0], Tmax = Tmax_p[0], Tmin = Tmin_p[0];
    const float dT   = temp - Tmax;
    const float melt = smooth_step(dT) * smooth_step(S_snow) *
                       fminf(S_snow, Df * dT);
    const float Ps  = smooth_step(Tmin - temp) * precp;
    const float Pr  = smooth_step(temp - Tmin) * precp;
    const float swv = smooth_step(S_water);

    const float dS1 = Ps - melt;
    const float dS2 = Pr + melt - swv * lday * __expf(ET) - swv * __expf(Q);

    if (i < n) {
        out[i]     = dS1;
        out[n + i] = dS2;
    }
}

extern "C" void kernel_launch(void* const* d_in, const int* in_sizes, int n_in,
                              void* d_out, int out_size, void* d_ws, size_t ws_size,
                              hipStream_t stream) {
    const float* t_in   = (const float*)d_in[0];
    const float* ssnow  = (const float*)d_in[1];
    const float* swater = (const float*)d_in[2];
    // d_in[3] = ts: uniform grid 0..T-1, index arithmetic replaces it
    const float* precp  = (const float*)d_in[4];
    const float* temp   = (const float*)d_in[5];
    const float* lday   = (const float*)d_in[6];
    const float* etW1 = (const float*)d_in[7];
    const float* etb1 = (const float*)d_in[8];
    const float* etW2 = (const float*)d_in[9];
    const float* etb2 = (const float*)d_in[10];
    const float* etW3 = (const float*)d_in[11];
    const float* etb3 = (const float*)d_in[12];
    const float* qW1  = (const float*)d_in[13];
    const float* qb1  = (const float*)d_in[14];
    const float* qW2  = (const float*)d_in[15];
    const float* qb2  = (const float*)d_in[16];
    const float* qW3  = (const float*)d_in[17];
    const float* qb3  = (const float*)d_in[18];
    const float* Df   = (const float*)d_in[19];
    const float* Tmax = (const float*)d_in[20];
    const float* Tmin = (const float*)d_in[21];

    const int n    = in_sizes[0];
    const int Tlen = in_sizes[4];
    const int blocks = (n + 255) / 256;

    m50_rhs_kernel<<<blocks, 256, 0, stream>>>(
        t_in, ssnow, swater, precp, temp, lday,
        etW1, etb1, etW2, etb2, etW3, etb3,
        qW1, qb1, qW2, qb2, qW3, qb3,
        Df, Tmax, Tmin,
        (float*)d_out, n, Tlen);
}
